// SparseTransformer_51719996179050
// MI455X (gfx1250) — matrix operand order repaired and checked by reading, not compiled, not run on hardware
//
#include <hip/hip_runtime.h>
#include <cmath>

// ---------- types ----------
typedef __attribute__((ext_vector_type(16))) __bf16   v16bf;
typedef __attribute__((ext_vector_type(8)))  __bf16   v8bf;
typedef __attribute__((ext_vector_type(8)))  float    v8f;
typedef __attribute__((ext_vector_type(4)))  unsigned v4u;
typedef __attribute__((ext_vector_type(8)))  int      v8i;
typedef __attribute__((ext_vector_type(4)))  int      v4i;

__device__ __forceinline__ __bf16 f2bf(float f) {
  union { float f; unsigned u; } v; v.f = f;
  unsigned r = v.u + 0x7FFFu + ((v.u >> 16) & 1u);   // round-to-nearest-even
  unsigned short h = (unsigned short)(r >> 16);
  return __builtin_bit_cast(__bf16, h);
}

// ---------------------------------------------------------------------------
// TDM: DMA one 128-row x 32-element bf16 tile (row stride = stride_elems) into
// LDS at byte offset lds_off. D# pad fields add 16B after every 64B row ->
// LDS row stride 80B (conflict-free for the WMMA fragment ds_load_b128s).
// Tracked on TENSORcnt; EXEC-independent, issued from one wave only.
// ---------------------------------------------------------------------------
__device__ __forceinline__ void tdm_load_tile_bf16(unsigned lds_off,
                                                   const __bf16* gp,
                                                   unsigned stride_elems)
{
  const unsigned long long ga = (unsigned long long)(uintptr_t)gp;
  v4u g0;
  g0[0] = 1u;                                          // count=1, user descriptor
  g0[1] = lds_off;                                     // lds_addr (bytes)
  g0[2] = (unsigned)(ga & 0xFFFFFFFFu);                // global_addr lo
  g0[3] = (unsigned)((ga >> 32) & 0x1FFFFFFu) | (2u << 30);  // addr hi | type=2
  const unsigned td0 = stride_elems;                   // tensor_dim0 (elems)
  const unsigned td1 = 0x40000000u;                    // tensor_dim1 (huge)
  v8i g1;
  g1[0] = (int)((1u << 16)      // data_size = 1 (2 bytes)
              | (1u << 20)      // pad_enable
              | (3u << 22)      // pad_interval: 16 DWORDs (one 64B row)
              | (3u << 25));    // pad_amount:   4 DWORDs (16B)
  g1[1] = (int)((td0 & 0xFFFFu) << 16);                         // dim0[15:0]
  g1[2] = (int)((td0 >> 16) | ((td1 & 0xFFFFu) << 16));         // dim0 hi | dim1 lo
  g1[3] = (int)(((td1 >> 16) & 0xFFFFu) | (32u << 16));         // dim1 hi | tile_dim0=32
  g1[4] = (int)128u;                                            // tile_dim1=128
  g1[5] = (int)stride_elems;                                    // dim0_stride lo32
  g1[6] = 0;
  g1[7] = 0;
  v4i z4; z4[0] = 0; z4[1] = 0; z4[2] = 0; z4[3] = 0;           // 2D: groups 2/3 unused
#if __clang_major__ >= 23
  v8i z8; z8[0]=0; z8[1]=0; z8[2]=0; z8[3]=0; z8[4]=0; z8[5]=0; z8[6]=0; z8[7]=0;
  __builtin_amdgcn_tensor_load_to_lds(g0, g1, z4, z4, z8, 0);
#else
  __builtin_amdgcn_tensor_load_to_lds(g0, g1, z4, z4, 0);
#endif
}

// ---------------------------------------------------------------------------
// WMMA GEMM (all-bf16 inputs):  C = (A[M,K] @ B[N,K]^T + bias) * scale
// OUT: 0 = f32 C, 1 = bf16 C, 2 = bf16 C stored transposed (C[n][m], ldc = M)
// 256 threads = 8 wave32; tile 128x128x32; TDM double-buffered staging.
// ---------------------------------------------------------------------------
template <int OUT>
__launch_bounds__(256)
__global__ void gemm_bt_wmma(const __bf16* __restrict__ A, const __bf16* __restrict__ B,
                             const float* __restrict__ bias, void* __restrict__ Cv,
                             int Kdim, int lda, int ldb, int ldc, float scale)
{
  constexpr int BM = 128, BN = 128, BK = 32, LP = 40;
  __shared__ __bf16 As[2][BM][LP];
  __shared__ __bf16 Bs[2][BN][LP];
  constexpr unsigned STAGE = BM * LP * (unsigned)sizeof(__bf16);   // 10240 B

  const int t    = threadIdx.x;
  const int lane = t & 31;
  const int wid  = t >> 5;
  const int wm   = wid & 3;         // wave M index -> 32 rows
  const int wn   = wid >> 2;        // wave N index -> 64 cols
  const int l16  = lane & 15;
  const int lh   = lane >> 4;
  const int rowBase = blockIdx.y * BM;
  const int colBase = blockIdx.x * BN;

  const __bf16* Ab = A + (size_t)rowBase * lda;
  const __bf16* Bb = B + (size_t)colBase * ldb;
  const unsigned aOff = (unsigned)(uintptr_t)&As[0][0][0];   // flat lo32 == LDS byte addr
  const unsigned bOff = (unsigned)(uintptr_t)&Bs[0][0][0];

  v8f acc[2][4];
  #pragma unroll
  for (int i = 0; i < 2; ++i)
    #pragma unroll
    for (int j = 0; j < 4; ++j)
      #pragma unroll
      for (int e = 0; e < 8; ++e) acc[i][j][e] = 0.0f;

  const bool issuer = (wid == 0);    // wave-uniform: one wave programs the TDM
  if (issuer) {
    tdm_load_tile_bf16(aOff, Ab, (unsigned)lda);
    tdm_load_tile_bf16(bOff, Bb, (unsigned)ldb);
  }

  int cur = 0;
  for (int k0 = 0; k0 < Kdim; k0 += BK) {
    if (issuer) __builtin_amdgcn_s_wait_tensorcnt(0);   // stage `cur` landed
    __syncthreads();                                    // visible to all waves; old buffer free
    if (issuer && (k0 + BK) < Kdim) {                   // prefetch next stage behind WMMAs
      const int nxt = cur ^ 1;
      tdm_load_tile_bf16(aOff + (unsigned)nxt * STAGE, Ab + (k0 + BK), (unsigned)lda);
      tdm_load_tile_bf16(bOff + (unsigned)nxt * STAGE, Bb + (k0 + BK), (unsigned)ldb);
    }

    // -------- fragments (ISA 7.12.2 layouts) --------
    v16bf afr[2], bfr[4];
    #pragma unroll
    for (int i = 0; i < 2; ++i) {
      const int r = wm * 32 + i * 16 + l16;
      v8bf lo = *(const v8bf*)&As[cur][r][lh * 8];        // K {0..7} / {8..15}
      v8bf hi = *(const v8bf*)&As[cur][r][16 + lh * 8];   // K {16..23} / {24..31}
      afr[i] = __builtin_shufflevector(lo, hi, 0, 1, 2, 3, 4, 5, 6, 7,
                                       8, 9, 10, 11, 12, 13, 14, 15);
    }
    #pragma unroll
    for (int j = 0; j < 4; ++j) {
      const int n = wn * 64 + j * 16 + l16;
      v8bf lo = *(const v8bf*)&Bs[cur][n][lh * 8];        // K {0..7} / {8..15}
      v8bf hi = *(const v8bf*)&Bs[cur][n][16 + lh * 8];   // K {16..23} / {24..31}
      bfr[j] = __builtin_shufflevector(lo, hi, 0, 1, 2, 3, 4, 5, 6, 7,
                                       8, 9, 10, 11, 12, 13, 14, 15);
    }

    // -------- 8x v_wmma_f32_16x16x32_bf16 --------
    #pragma unroll
    for (int i = 0; i < 2; ++i)
      #pragma unroll
      for (int j = 0; j < 4; ++j)
        acc[i][j] = __builtin_amdgcn_wmma_f32_16x16x32_bf16(
            false, afr[i], false, bfr[j], (short)0, acc[i][j], false, false);

    cur ^= 1;
  }

  // -------- epilogue --------
  #pragma unroll
  for (int i = 0; i < 2; ++i) {
    const int r0 = rowBase + wm * 32 + i * 16 + lh * 8;
    #pragma unroll
    for (int j = 0; j < 4; ++j) {
      const int c = colBase + wn * 64 + j * 16 + l16;
      const float bb = bias ? bias[c] : 0.0f;
      if (OUT == 0) {
        float* Cf = (float*)Cv;
        #pragma unroll
        for (int r = 0; r < 8; ++r)
          Cf[(size_t)(r0 + r) * ldc + c] = (acc[i][j][r] + bb) * scale;
      } else if (OUT == 1) {
        __bf16* Cb = (__bf16*)Cv;
        #pragma unroll
        for (int r = 0; r < 8; ++r)
          Cb[(size_t)(r0 + r) * ldc + c] = f2bf((acc[i][j][r] + bb) * scale);
      } else {                       // bf16, transposed: contiguous 16B per lane
        __bf16* Ct = (__bf16*)Cv;
        v8bf p;
        #pragma unroll
        for (int r = 0; r < 8; ++r) p[r] = f2bf((acc[i][j][r] + bb) * scale);
        *(v8bf*)&Ct[(size_t)c * ldc + r0] = p;
      }
    }
  }
}

// ---------------------------------------------------------------------------
// Sparsemax, one block per row, sort-free (bisection + exact support pass).
// Writes f32 attn in-place (d_out) and a bf16 copy for the attn@V GEMM.
// ---------------------------------------------------------------------------
__launch_bounds__(256)
__global__ void sparsemax_kernel(float* __restrict__ z_glob, __bf16* __restrict__ zb_glob,
                                 int Fdim)
{
  __shared__ float row[8192];
  __shared__ float red[256];
  __shared__ float red2[256];

  const int t = threadIdx.x;
  float*  z  = z_glob  + (size_t)blockIdx.x * Fdim;
  __bf16* zb = zb_glob + (size_t)blockIdx.x * Fdim;

  float lmax = -3.0e38f;
  for (int i = t; i < Fdim; i += 256) { float v = z[i]; row[i] = v; lmax = fmaxf(lmax, v); }
  red[t] = lmax; __syncthreads();
  for (int s = 128; s > 0; s >>= 1) {
    if (t < s) red[t] = fmaxf(red[t], red[t + s]);
    __syncthreads();
  }
  const float zmax = red[0]; __syncthreads();

  float lo = zmax - 1.0f, hi = zmax;          // f(lo) >= 1, f(hi) = 0
  for (int it = 0; it < 26; ++it) {
    const float mid = 0.5f * (lo + hi);
    float ls = 0.0f;
    for (int i = t; i < Fdim; i += 256) ls += fmaxf(row[i] - mid, 0.0f);
    red[t] = ls; __syncthreads();
    for (int s = 128; s > 0; s >>= 1) {
      if (t < s) red[t] += red[t + s];
      __syncthreads();
    }
    const float ssum = red[0]; __syncthreads();
    if (ssum >= 1.0f) lo = mid; else hi = mid;   // uniform branch
  }

  float ls = 0.0f, lk = 0.0f;                  // exact tau over support {z > lo}
  for (int i = t; i < Fdim; i += 256) {
    const float v = row[i];
    if (v > lo) { ls += v; lk += 1.0f; }
  }
  red[t] = ls; red2[t] = lk; __syncthreads();
  for (int s = 128; s > 0; s >>= 1) {
    if (t < s) { red[t] += red[t + s]; red2[t] += red2[t + s]; }
    __syncthreads();
  }
  const float tau = (red[0] - 1.0f) / red2[0];

  for (int i = t; i < Fdim; i += 256) {
    const float a = fmaxf(row[i] - tau, 0.0f);
    z[i]  = a;
    zb[i] = f2bf(a);
  }
}

// ---------------------------------------------------------------------------
__launch_bounds__(256)
__global__ void cvt_bf16_kernel(const float* __restrict__ in, __bf16* __restrict__ out, int n)
{
  const int i = (blockIdx.x * 256 + threadIdx.x) * 8;
  if (i >= n) return;
  const float4 a = *(const float4*)(in + i);
  const float4 b = *(const float4*)(in + i + 4);
  v8bf p;
  p[0] = f2bf(a.x); p[1] = f2bf(a.y); p[2] = f2bf(a.z); p[3] = f2bf(a.w);
  p[4] = f2bf(b.x); p[5] = f2bf(b.y); p[6] = f2bf(b.z); p[7] = f2bf(b.w);
  *(v8bf*)(out + i) = p;
}

__global__ void copy4_kernel(const float4* __restrict__ in, float4* __restrict__ out, int n4)
{
  const int i = blockIdx.x * blockDim.x + threadIdx.x;
  if (i < n4) out[i] = in[i];
}

// ---------------------------------------------------------------------------
extern "C" void kernel_launch(void* const* d_in, const int* in_sizes, int n_in,
                              void* d_out, int out_size, void* d_ws, size_t ws_size,
                              hipStream_t stream)
{
  const float* x  = (const float*)d_in[0];  // [B, D]
  const float* Xd = (const float*)d_in[1];  // [F, D]
  const float* Wq = (const float*)d_in[2];  // [M, D]
  const float* bq = (const float*)d_in[3];
  const float* Wk = (const float*)d_in[4];  // [M, D]
  const float* bk = (const float*)d_in[5];
  const float* Wv = (const float*)d_in[6];  // [D, D]
  const float* bv = (const float*)d_in[7];

  const int Mdim = in_sizes[3];             // 256
  const int D    = in_sizes[7];             // 1024
  const int B    = in_sizes[0] / D;         // 4096
  const int F    = in_sizes[1] / D;         // 8192
  const float scale = 1.0f / sqrtf((float)Mdim);

  float* out      = (float*)d_out;
  float* out_xhat = out + (size_t)B * D;
  float* attn     = out + 2 * (size_t)B * D;   // [B, F]: scores -> sparsemax in-place

  // bf16 workspace
  __bf16* ws      = (__bf16*)d_ws;
  __bf16* x_bf    = ws;                              // B*D
  __bf16* X_bf    = x_bf  + (size_t)B * D;           // F*D
  __bf16* Wq_bf   = X_bf  + (size_t)F * D;           // M*D
  __bf16* Wk_bf   = Wq_bf + (size_t)Mdim * D;        // M*D
  __bf16* Wv_bf   = Wk_bf + (size_t)Mdim * D;        // D*D
  __bf16* Q_bf    = Wv_bf + (size_t)D * D;           // B*M
  __bf16* K_bf    = Q_bf  + (size_t)B * Mdim;        // F*M
  __bf16* Vt_bf   = K_bf  + (size_t)F * Mdim;        // D*F (V transposed)
  __bf16* attn_bf = Vt_bf + (size_t)D * F;           // B*F

  const dim3 blk(256);
  auto cvt_grid = [](size_t n) { return dim3((unsigned)((n / 8 + 255) / 256)); };

  // f32 -> bf16 once (bandwidth-trivial)
  cvt_bf16_kernel<<<cvt_grid((size_t)B * D),    blk, 0, stream>>>(x,  x_bf,  B * D);
  cvt_bf16_kernel<<<cvt_grid((size_t)F * D),    blk, 0, stream>>>(Xd, X_bf,  F * D);
  cvt_bf16_kernel<<<cvt_grid((size_t)Mdim * D), blk, 0, stream>>>(Wq, Wq_bf, Mdim * D);
  cvt_bf16_kernel<<<cvt_grid((size_t)Mdim * D), blk, 0, stream>>>(Wk, Wk_bf, Mdim * D);
  cvt_bf16_kernel<<<cvt_grid((size_t)D * D),    blk, 0, stream>>>(Wv, Wv_bf, D * D);

  // Q = (x @ Wq^T + bq) * scale   -> bf16
  gemm_bt_wmma<1><<<dim3(Mdim / 128, B / 128), blk, 0, stream>>>(
      x_bf, Wq_bf, bq, Q_bf, D, D, D, Mdim, scale);
  // K = (X @ Wk^T + bk) * scale   -> bf16
  gemm_bt_wmma<1><<<dim3(Mdim / 128, F / 128), blk, 0, stream>>>(
      X_bf, Wk_bf, bk, K_bf, D, D, D, Mdim, scale);
  // V^T = (X @ Wv^T + bv)^T       -> bf16 transposed store [D, F]
  gemm_bt_wmma<2><<<dim3(D / 128, F / 128), blk, 0, stream>>>(
      X_bf, Wv_bf, bv, Vt_bf, D, D, D, F, 1.0f);
  // scores = Q @ K^T              -> f32 into attn slot of d_out
  gemm_bt_wmma<0><<<dim3(F / 128, B / 128), blk, 0, stream>>>(
      Q_bf, K_bf, nullptr, attn, Mdim, Mdim, Mdim, F, 1.0f);
  // sparsemax in-place (+ bf16 copy)
  sparsemax_kernel<<<dim3(B), blk, 0, stream>>>(attn, attn_bf, F);
  // x_hat = attn @ (V^T)^T        -> f32
  gemm_bt_wmma<0><<<dim3(D / 128, B / 128), blk, 0, stream>>>(
      attn_bf, Vt_bf, nullptr, out_xhat, F, F, F, D, 1.0f);
  // pass-through x
  const int n4 = (B * D) / 4;
  copy4_kernel<<<dim3((n4 + 255) / 256), blk, 0, stream>>>(
      (const float4*)x, (float4*)out, n4);
}
